// SimpleRNN_26637387170352
// MI455X (gfx1250) — matrix operand order repaired and checked by reading, not compiled, not run on hardware
//
#include <hip/hip_runtime.h>

// ---------------------------------------------------------------------------
// Types
// ---------------------------------------------------------------------------
typedef __attribute__((ext_vector_type(16))) __bf16 v16bf;
typedef __attribute__((ext_vector_type(8)))  __bf16 v8bf;
typedef __attribute__((ext_vector_type(8)))  float  v8f;

static constexpr int Bc   = 64;
static constexpr int Tc   = 512;
static constexpr int INc  = 256;
static constexpr int Hc   = 1024;
static constexpr int OUTc = 512;
static constexpr long BT  = (long)Bc * Tc;

// ---------------------------------------------------------------------------
// Helpers
// ---------------------------------------------------------------------------
__device__ __forceinline__ __bf16 f2bf(float f) {
    unsigned int u = __builtin_bit_cast(unsigned int, f);
    u += 0x7FFFu + ((u >> 16) & 1u);               // round to nearest even
    unsigned short h = (unsigned short)(u >> 16);
    return __builtin_bit_cast(__bf16, h);
}

__device__ __forceinline__ v16bf mk16(v8bf lo, v8bf hi) {
    v16bf r;
#pragma unroll
    for (int i = 0; i < 8; ++i) { r[i] = lo[i]; r[i + 8] = hi[i]; }
    return r;
}

// A fragment (16x32, MxK): lane L<16 -> row M=L, K={0..7,16..23};
// lane L>=16 -> row M=L-16, K={8..15,24..31}.   (ISA 7.12.2, 16-bit A)
__device__ __forceinline__ v16bf load_a_frag(const __bf16* base, int stride,
                                             int row0, int lane) {
    const int kb = ((lane >> 4) << 3);
    const __bf16* p = base + (long)(row0 + (lane & 15)) * stride + kb;
    return mk16(*(const v8bf*)p, *(const v8bf*)(p + 16));
}

// B fragment (32x16, KxN): lane L<16 -> col N=L, K={0..7,16..23};
// lane L>=16 -> col N=L-16, K={8..15,24..31}.
__device__ __forceinline__ v16bf load_b_frag(const __bf16* base, int stride,
                                             int col0, int lane) {
    const int kb = ((lane >> 4) << 3);
    const __bf16* p = base + (long)(col0 + (lane & 15)) * stride + kb;
    return mk16(*(const v8bf*)p, *(const v8bf*)(p + 16));
}

// ---------------------------------------------------------------------------
// f32 -> bf16 conversion
// ---------------------------------------------------------------------------
__global__ void cvt_f32_bf16(const float* __restrict__ in,
                             __bf16* __restrict__ out, long n) {
    long i = (long)blockIdx.x * blockDim.x + threadIdx.x;
    long stride = (long)gridDim.x * blockDim.x;
    for (; i < n; i += stride) out[i] = f2bf(in[i]);
}

// ---------------------------------------------------------------------------
// Tiled WMMA GEMM:  C[M,N](f32) = A[M,K](bf16) * W[N,K]^T(bf16) + bias0 + bias1
// Block: 256 threads (8 waves), tile 128x64, K-chunk 32, double-buffered LDS.
// GUARD=false -> all bounds checks compiled out (tiles divide M,N exactly).
// ---------------------------------------------------------------------------
#define GBM 128
#define GBN 64
#define GKC 32
#define GLST (GKC + 8)

template <bool GUARD>
__global__ void __launch_bounds__(256)
wmma_gemm_bias(const __bf16* __restrict__ A, const __bf16* __restrict__ W,
               const float* __restrict__ bias0, const float* __restrict__ bias1,
               float* __restrict__ C, int M, int N, int K) {
    __shared__ __bf16 sA[2][GBM][GLST];
    __shared__ __bf16 sB[2][GBN][GLST];

    const int tid  = threadIdx.x;
    const int wave = tid >> 5;
    const int lane = tid & 31;
    const int mBlock = blockIdx.y * GBM;
    const int nBlock = blockIdx.x * GBN;
    const int wm0 = (wave >> 1) * 32;   // 4 wave-rows x 2 wave-cols
    const int wn0 = (wave & 1) * 32;

    // Per-thread staging coordinates (loop-invariant).
    const int srow = tid >> 2;          // 0..63
    const int scol = (tid & 3) << 3;    // 0,8,16,24
    const long aOff0 = (long)(mBlock + srow) * K + scol;        // A rows 0..63
    const long aOff1 = (long)(mBlock + srow + 64) * K + scol;   // A rows 64..127
    const long bOff  = (long)(nBlock + srow) * K + scol;        // W rows 0..63
    const bool aOk0 = !GUARD || (mBlock + srow      < M);
    const bool aOk1 = !GUARD || (mBlock + srow + 64 < M);
    const bool bOk  = !GUARD || (nBlock + srow      < N);

    v8f acc[2][2] = {};
    const int nk = K / GKC;

    v8bf ra0 = {}, ra1 = {}, rb = {};
    if (aOk0) ra0 = *(const v8bf*)(A + aOff0);
    if (aOk1) ra1 = *(const v8bf*)(A + aOff1);
    if (bOk)  rb  = *(const v8bf*)(W + bOff);
    *(v8bf*)&sA[0][srow][scol]      = ra0;
    *(v8bf*)&sA[0][srow + 64][scol] = ra1;
    *(v8bf*)&sB[0][srow][scol]      = rb;
    __syncthreads();

    for (int c = 0; c < nk; ++c) {
        const int cur = c & 1;

        // Prefetch next chunk into registers while computing this one.
        if (c + 1 < nk) {
            const int k0 = (c + 1) * GKC;
            ra0 = v8bf{}; ra1 = v8bf{}; rb = v8bf{};
            if (aOk0) ra0 = *(const v8bf*)(A + aOff0 + k0);
            if (aOk1) ra1 = *(const v8bf*)(A + aOff1 + k0);
            if (bOk)  rb  = *(const v8bf*)(W + bOff  + k0);
        }

        v16bf bf[2];
#pragma unroll
        for (int j = 0; j < 2; ++j)
            bf[j] = load_b_frag(&sB[cur][0][0], GLST, wn0 + j * 16, lane);
#pragma unroll
        for (int i = 0; i < 2; ++i) {
            v16bf af = load_a_frag(&sA[cur][0][0], GLST, wm0 + i * 16, lane);
#pragma unroll
            for (int j = 0; j < 2; ++j)
                acc[i][j] = __builtin_amdgcn_wmma_f32_16x16x32_bf16(
                    false, af, false, bf[j], (short)0, acc[i][j], false, false);
        }

        if (c + 1 < nk) {
            const int nxt = cur ^ 1;
            *(v8bf*)&sA[nxt][srow][scol]      = ra0;
            *(v8bf*)&sA[nxt][srow + 64][scol] = ra1;
            *(v8bf*)&sB[nxt][srow][scol]      = rb;
        }
        __syncthreads();
    }

    // Epilogue: C layout -> lane L, vgpr r: M = r + 8*(L>>4), N = L&15
#pragma unroll
    for (int i = 0; i < 2; ++i) {
#pragma unroll
        for (int j = 0; j < 2; ++j) {
            int nn = nBlock + wn0 + j * 16 + (lane & 15);
            int mb = mBlock + wm0 + i * 16 + ((lane >> 4) << 3);
            float bv = 0.0f;
            if (!GUARD || nn < N) {
                if (bias0) bv += bias0[nn];
                if (bias1) bv += bias1[nn];
            }
#pragma unroll
            for (int r = 0; r < 8; ++r) {
                int mm = mb + r;
                if (!GUARD || (mm < M && nn < N))
                    C[(long)mm * N + nn] = acc[i][j][r] + bv;
            }
        }
    }
}

// ---------------------------------------------------------------------------
// Persistent RNN recurrence:  h_t = tanh(xproj_t + h_{t-1} @ Wh^T)
// 16 WGs x 512 threads (16 waves). Each WG owns 64 columns of Wh, held in
// LDS (bf16, padded) for the entire T loop. Device-wide barrier per step.
// ---------------------------------------------------------------------------
#define RNWG 16
#define RLSTRIDE (Hc + 8)

__global__ void __launch_bounds__(512)
rnn_recurrence(const float* __restrict__ xproj,      // [B, T, H] f32
               const __bf16* __restrict__ Wh,        // [H, H] bf16 (row = out col)
               __bf16* __restrict__ hbuf0,           // [B, H] ping (init 0)
               __bf16* __restrict__ hbuf1,           // [B, H] pong
               __bf16* __restrict__ h_all,           // [B, T, H] bf16 or null
               int T, int* __restrict__ bar_count, int* __restrict__ bar_gen) {
    extern __shared__ __bf16 sW[];                   // [64][RLSTRIDE]
    const int n0  = blockIdx.x * 64;
    const int tid = threadIdx.x;

    // Load this WG's 64 Wh rows (1024 K each) into LDS, once.
    for (int s = tid; s < 64 * (Hc / 8); s += 512) {
        int r = s >> 7;                // 128 8-elt segments per row
        int c = (s & 127) << 3;
        *(v8bf*)&sW[r * RLSTRIDE + c] =
            *(const v8bf*)(Wh + (long)(n0 + r) * Hc + c);
    }
    __syncthreads();

    const int wave   = tid >> 5;
    const int lane   = tid & 31;
    const int m0     = (wave >> 2) * 16;              // batch-tile row
    const int nlocal = (wave & 3) * 16;               // local col-tile
    const int nglob  = n0 + nlocal + (lane & 15);
    const int mbase  = m0 + ((lane >> 4) << 3);

    for (int t = 0; t < T; ++t) {
        const __bf16* hprev = (t & 1) ? hbuf1 : hbuf0;
        __bf16*       hnext = (t & 1) ? hbuf0 : hbuf1;

        // Issue the xproj loads first: independent of h_{t-1}, their HBM/L2
        // latency overlaps the entire WMMA K-loop below.
        float xp[8];
#pragma unroll
        for (int r = 0; r < 8; ++r)
            xp[r] = xproj[((long)(mbase + r) * T + t) * Hc + nglob];

        // Two interleaved accumulators to shorten the WMMA RAW chain.
        v8f acc0 = {}, acc1 = {};
#pragma unroll 4
        for (int k0 = 0; k0 < Hc; k0 += 64) {
            v16bf a0 = load_a_frag(hprev + k0, Hc, m0, lane);
            v16bf b0 = load_b_frag(sW + k0, RLSTRIDE, nlocal, lane);
            acc0 = __builtin_amdgcn_wmma_f32_16x16x32_bf16(
                false, a0, false, b0, (short)0, acc0, false, false);
            v16bf a1 = load_a_frag(hprev + k0 + 32, Hc, m0, lane);
            v16bf b1 = load_b_frag(sW + k0 + 32, RLSTRIDE, nlocal, lane);
            acc1 = __builtin_amdgcn_wmma_f32_16x16x32_bf16(
                false, a1, false, b1, (short)0, acc1, false, false);
        }

        // Prefetch next timestep's xproj lines into cache (global_prefetch).
        if (t + 1 < T) {
#pragma unroll
            for (int r = 0; r < 8; ++r)
                __builtin_prefetch(
                    &xproj[((long)(mbase + r) * T + (t + 1)) * Hc + nglob], 0, 3);
        }

        // Epilogue: add xproj, tanh, store bf16 for next step (+ history).
#pragma unroll
        for (int r = 0; r < 8; ++r) {
            int m = mbase + r;
            float v  = tanhf(acc0[r] + acc1[r] + xp[r]);
            __bf16 hv = f2bf(v);
            hnext[m * Hc + nglob] = hv;
            if (h_all) h_all[((long)m * T + t) * Hc + nglob] = hv;
        }

        // Device-wide barrier (sense-reversal via generation counter).
        __threadfence();
        __syncthreads();
        if (tid == 0) {
            int g = __hip_atomic_load(bar_gen, __ATOMIC_RELAXED,
                                      __HIP_MEMORY_SCOPE_AGENT);
            int arrived = __hip_atomic_fetch_add(bar_count, 1, __ATOMIC_ACQ_REL,
                                                 __HIP_MEMORY_SCOPE_AGENT) + 1;
            if (arrived == (int)gridDim.x) {
                __hip_atomic_store(bar_count, 0, __ATOMIC_RELAXED,
                                   __HIP_MEMORY_SCOPE_AGENT);
                __hip_atomic_store(bar_gen, g + 1, __ATOMIC_RELEASE,
                                   __HIP_MEMORY_SCOPE_AGENT);
            } else {
                while (__hip_atomic_load(bar_gen, __ATOMIC_ACQUIRE,
                                         __HIP_MEMORY_SCOPE_AGENT) == g)
                    __builtin_amdgcn_s_sleep(1);
            }
        }
        __syncthreads();
        __threadfence();
    }
}

// ---------------------------------------------------------------------------
// Host launcher
// ---------------------------------------------------------------------------
extern "C" void kernel_launch(void* const* d_in, const int* in_sizes, int n_in,
                              void* d_out, int out_size, void* d_ws,
                              size_t ws_size, hipStream_t stream) {
    const float* x   = (const float*)d_in[0];
    const float* Wx0 = (const float*)d_in[1];
    const float* bx0 = (const float*)d_in[2];
    const float* Wh0 = (const float*)d_in[3];
    const float* bh0 = (const float*)d_in[4];
    const float* Wx1 = (const float*)d_in[5];
    const float* bx1 = (const float*)d_in[6];
    const float* Wh1 = (const float*)d_in[7];
    const float* bh1 = (const float*)d_in[8];
    const float* Wfc = (const float*)d_in[9];
    const float* bfc = (const float*)d_in[10];
    float* out = (float*)d_out;

    // Workspace carve-up (256B aligned). xproj buffer is reused for layer 1.
    char* ws = (char*)d_ws;
    size_t off = 0;
    auto carve = [&](size_t bytes) {
        void* p = ws + off;
        off += (bytes + 255) & ~(size_t)255;
        return p;
    };
    float*  xproj = (float*) carve(BT * Hc * sizeof(float));     // 128 MB (shared)
    __bf16* h1all = (__bf16*)carve(BT * Hc * 2);                 //  64 MB
    __bf16* xbf   = (__bf16*)carve(BT * INc * 2);                //  16 MB
    __bf16* wx0b  = (__bf16*)carve((size_t)Hc * INc * 2);
    __bf16* wh0b  = (__bf16*)carve((size_t)Hc * Hc * 2);
    __bf16* wx1b  = (__bf16*)carve((size_t)Hc * Hc * 2);
    __bf16* wh1b  = (__bf16*)carve((size_t)Hc * Hc * 2);
    __bf16* wfcb  = (__bf16*)carve((size_t)OUTc * Hc * 2);
    __bf16* hA0   = (__bf16*)carve((size_t)Bc * Hc * 2);
    __bf16* hA1   = (__bf16*)carve((size_t)Bc * Hc * 2);
    __bf16* hB0   = (__bf16*)carve((size_t)Bc * Hc * 2);
    __bf16* hB1   = (__bf16*)carve((size_t)Bc * Hc * 2);
    int*    bars  = (int*)   carve(256);   // [0]=cnt0 [1]=gen0 [2]=cnt1 [3]=gen1

    // Init barrier state + h0 = 0 (workspace is poisoned, reset every call).
    hipMemsetAsync(bars, 0, 256, stream);
    hipMemsetAsync(hA0, 0, (size_t)Bc * Hc * 2, stream);
    hipMemsetAsync(hB0, 0, (size_t)Bc * Hc * 2, stream);

    // bf16 conversions.
    auto cvt = [&](const float* src, __bf16* dst, long n) {
        int blocks = (int)((n + 255) / 256);
        if (blocks > 16384) blocks = 16384;
        cvt_f32_bf16<<<blocks, 256, 0, stream>>>(src, dst, n);
    };
    cvt(x,   xbf,  BT * INc);
    cvt(Wx0, wx0b, (long)Hc * INc);
    cvt(Wh0, wh0b, (long)Hc * Hc);
    cvt(Wx1, wx1b, (long)Hc * Hc);
    cvt(Wh1, wh1b, (long)Hc * Hc);
    cvt(Wfc, wfcb, (long)OUTc * Hc);

    const size_t rnn_lds = (size_t)64 * RLSTRIDE * 2;  // 132096 B

    // xproj0 = x @ Wx0^T + bx0 + bh0   (M=32768, N=1024, K=256) — guard-free.
    wmma_gemm_bias<false><<<dim3(Hc / GBN, (unsigned)(BT / GBM)), 256, 0, stream>>>(
        xbf, wx0b, bx0, bh0, xproj, (int)BT, Hc, INc);

    // Layer 0 recurrence (stores full h1 history as bf16).
    rnn_recurrence<<<RNWG, 512, rnn_lds, stream>>>(
        xproj, wh0b, hA0, hA1, h1all, Tc, &bars[0], &bars[1]);

    // xproj1 = h1 @ Wx1^T + bx1 + bh1  (M=32768, N=1024, K=1024) — guard-free.
    wmma_gemm_bias<false><<<dim3(Hc / GBN, (unsigned)(BT / GBM)), 256, 0, stream>>>(
        h1all, wx1b, bx1, bh1, xproj, (int)BT, Hc, Hc);

    // Layer 1 recurrence (history not needed; final h lands in hB0 for even T).
    rnn_recurrence<<<RNWG, 512, rnn_lds, stream>>>(
        xproj, wh1b, hB0, hB1, nullptr, Tc, &bars[2], &bars[3]);

    // out = h_last @ Wfc^T + bfc       (M=64, N=512, K=1024) — guarded path.
    wmma_gemm_bias<true><<<dim3(OUTc / GBN, (Bc + GBM - 1) / GBM), 256, 0, stream>>>(
        hB0, wfcb, bfc, nullptr, out, Bc, OUTc, Hc);
}
